// TestModel_49366354100542
// MI455X (gfx1250) — matrix operand order repaired and checked by reading, not compiled, not run on hardware
//
#include <hip/hip_runtime.h>
#include <stddef.h>
#include <stdint.h>

// CDNA5 (gfx1250) wave32 WMMA types
typedef __attribute__((ext_vector_type(16))) __bf16       v16bf;
typedef __attribute__((ext_vector_type(8)))  float        v8f;
typedef __attribute__((ext_vector_type(8)))  unsigned int v8u;

#define WMMA_BF16(A, B, C) \
  __builtin_amdgcn_wmma_f32_16x16x32_bf16(false, (A), false, (B), (short)0, (C), false, false)

// Problem sizes (fixed by the reference)
// b=32, i=2048, j=64, n_in=16, n_out=32; P = i/2 = 1024 i-pairs (K=32 packs 2 i x 16 n)

// Workspace layout (bytes)
#define OFF_WB 0UL                    // bf16 W fragments: 64*1024*2 frags * 1KB = 128 MB
#define OFF_XA 134217728UL            // bf16 x fragments (A-layout): 2 MB
#define OFF_XB 136314880UL            // bf16 x fragments (B-layout): 2 MB
#define OFF_T  138412032UL            // logits t[j][i][b] f32: 16 MB
#define OFF_C  155189248UL            // coupling c[j][i][b] f32: 16 MB
#define OFF_SP 171966464UL            // partial s [chunk=8][b=32][j=64][m=32] f32: 2 MB
#define OFF_VS 174063616UL            // vsum [b][j][m] f32: 256 KB
#define WS_NEED 174325760UL

static __device__ __forceinline__ v8f vzero8f() {
  v8f z;
#pragma unroll
  for (int i = 0; i < 8; ++i) z[i] = 0.0f;
  return z;
}

static __device__ __forceinline__ v16bf vzero16bf() {
  v16bf z;
#pragma unroll
  for (int i = 0; i < 16; ++i) z[i] = (__bf16)0.0f;
  return z;
}

// Involution mapping a B-layout 16-bit fragment (lanes0-15:K0-15, lanes16-31:K16-31)
// to the A-layout (K = e + 8*((e>=8)+(lane>=16))) and vice versa.
static __device__ __forceinline__ v16bf swap_halves(v16bf f, int lane) {
  v8u a = __builtin_bit_cast(v8u, f);
  v8u s;
#pragma unroll
  for (int i = 0; i < 8; ++i) s[i] = (unsigned)__shfl_xor((int)a[i], 16, 32);
  v8u g;
  if (lane < 16) {
    g[0] = a[0]; g[1] = a[1]; g[2] = a[2]; g[3] = a[3];
    g[4] = s[0]; g[5] = s[1]; g[6] = s[2]; g[7] = s[3];
  } else {
    g[0] = s[4]; g[1] = s[5]; g[2] = s[6]; g[3] = s[7];
    g[4] = a[4]; g[5] = a[5]; g[6] = a[6]; g[7] = a[7];
  }
  return __builtin_bit_cast(v16bf, g);
}

// --- W f32 -> bf16 B-layout fragments.  One wave per fragment.
// frag id = (j*1024+p)*2+mt ; lane(l,h): value[e] = W[j, 2p+h, mt*16+l, e]
__global__ void caps_convW(const float* __restrict__ W, __bf16* __restrict__ WbB) {
  const int wave = (blockIdx.x * blockDim.x + threadIdx.x) >> 5;
  const int lane = threadIdx.x & 31;
  const int l = lane & 15, h = lane >> 4;
  const int mt = wave & 1;
  const int p  = (wave >> 1) & 1023;
  const int j  = wave >> 11;
  const float* src = W + (((size_t)j * 2048 + 2 * p + h) * 32 + mt * 16 + l) * 16;
  v16bf f;
#pragma unroll
  for (int e = 0; e < 16; ++e) f[e] = (__bf16)src[e];
  *(v16bf*)(WbB + (size_t)wave * 512 + (size_t)lane * 16) = f;
}

// --- x f32 -> bf16 fragments in BOTH layouts. frag id = bt*1024+p.
__global__ void caps_convX(const float* __restrict__ x,
                           __bf16* __restrict__ XbA, __bf16* __restrict__ XbB) {
  const int wave = (blockIdx.x * blockDim.x + threadIdx.x) >> 5;
  const int lane = threadIdx.x & 31;
  const int l = lane & 15, h = lane >> 4;
  const int bt = wave >> 10;
  const int p  = wave & 1023;
  const int b  = bt * 16 + l;
  const float* x0 = x + ((size_t)b * 2048 + 2 * p) * 16;
  const float* x1 = x0 + 16;
  v16bf fa, fb;
#pragma unroll
  for (int e = 0; e < 8; ++e) {
    fa[e]     = (__bf16)x0[e + 8 * h];   // A-layout: slots 0-7 = i0
    fa[e + 8] = (__bf16)x1[e + 8 * h];   //           slots 8-15 = i1
  }
  const float* xi = h ? x1 : x0;         // B-layout: lane half selects i
#pragma unroll
  for (int e = 0; e < 16; ++e) fb[e] = (__bf16)xi[e];
  *(v16bf*)(XbA + (size_t)wave * 512 + (size_t)lane * 16) = fa;
  *(v16bf*)(XbB + (size_t)wave * 512 + (size_t)lane * 16) = fb;
}

// --- s-pass: s_j[b,m] = sum_i c[b,j,i] * u_hat[b,j,i,m]  via K=(i,n) WMMA chain.
// grid: 512 blocks = j(64) x chunk(8); 256 threads = 8 waves x 16 pairs each.
__global__ void caps_spass(const __bf16* __restrict__ WbB, const __bf16* __restrict__ XbA,
                           const float* __restrict__ cb, const int uniformC,
                           float* __restrict__ spart) {
  __shared__ float red[8][1024];
  const int j = blockIdx.x >> 3, chunk = blockIdx.x & 7;
  const int w = threadIdx.x >> 5, lane = threadIdx.x & 31;
  const int l = lane & 15, h = lane >> 4;
  v8f C[2][2];
#pragma unroll
  for (int bt = 0; bt < 2; ++bt)
#pragma unroll
    for (int mt = 0; mt < 2; ++mt) C[bt][mt] = vzero8f();

  const int pbase = chunk * 128 + w * 16;
  for (int pp = 0; pp < 16; ++pp) {
    const int p = pbase + pp;
    const __bf16* wb = WbB + ((size_t)(j * 1024 + p) * 2) * 512 + (size_t)lane * 16;
    const v16bf B0 = swap_halves(*(const v16bf*)(wb), lane);
    const v16bf B1 = swap_halves(*(const v16bf*)(wb + 512), lane);
#pragma unroll
    for (int bt = 0; bt < 2; ++bt) {
      const v16bf xf = *(const v16bf*)(XbA + ((size_t)(bt * 1024 + p)) * 512 + (size_t)lane * 16);
      float c0, c1;
      if (uniformC) {
        c0 = 0.015625f; c1 = 0.015625f;           // softmax of zeros over j=64
      } else {
        const int b = bt * 16 + l;
        c0 = cb[((size_t)j * 2048 + 2 * p    ) * 32 + b];
        c1 = cb[((size_t)j * 2048 + 2 * p + 1) * 32 + b];
      }
      v16bf a;                                     // fold c into bf16 A operand
#pragma unroll
      for (int e = 0; e < 8; ++e) a[e]     = (__bf16)((float)xf[e]     * c0);
#pragma unroll
      for (int e = 0; e < 8; ++e) a[e + 8] = (__bf16)((float)xf[e + 8] * c1);
      C[bt][0] = WMMA_BF16(a, B0, C[bt][0]);
      C[bt][1] = WMMA_BF16(a, B1, C[bt][1]);
    }
  }
  // cross-wave reduce in LDS, then plain (non-atomic) partial store
#pragma unroll
  for (int bt = 0; bt < 2; ++bt)
#pragma unroll
    for (int mt = 0; mt < 2; ++mt)
#pragma unroll
      for (int r = 0; r < 8; ++r) {
        const int b = bt * 16 + r + 8 * h;         // C-tile rows = b
        const int m = mt * 16 + l;                 // C-tile cols = m
        red[w][b * 32 + m] = C[bt][mt][r];
      }
  __syncthreads();
  for (int e = threadIdx.x; e < 1024; e += 256) {
    float s = 0.0f;
#pragma unroll
    for (int ww = 0; ww < 8; ++ww) s += red[ww][e];
    const int b = e >> 5, m = e & 31;
    spart[(((size_t)chunk * 32 + b) * 64 + j) * 32 + m] = s;
  }
}

// --- squash: v = s*(n2/(1+n2))/sqrt(n2+eps); v -> d_out, vsum (+)= v
__global__ void caps_squash(const float* __restrict__ spart, const float* __restrict__ bias,
                            float* __restrict__ out, float* __restrict__ vsum, const int first) {
  const int b = blockIdx.x >> 6, j = blockIdx.x & 63;
  const int m = threadIdx.x;                       // 32 threads = one wave
  float s = bias[j * 32 + m];
#pragma unroll
  for (int ch = 0; ch < 8; ++ch)
    s += spart[(((size_t)ch * 32 + b) * 64 + j) * 32 + m];
  float n2 = s * s;
#pragma unroll
  for (int off = 16; off > 0; off >>= 1) n2 += __shfl_xor(n2, off, 32);
  const float v = s * (n2 / (1.0f + n2)) * rsqrtf(n2 + 1e-8f);
  const size_t o = ((size_t)b * 64 + j) * 32 + m;
  out[o]  = v;
  vsum[o] = first ? v : (vsum[o] + v);
}

// --- t-pass: t[b,j,i] = <vsum[b,j,:], u_hat[b,j,i,:]> ; transposed tiles D[m,b]
// so the m-contraction is 8 in-lane FMAs. grid: 1024 = j(64) x chunk(16); 128 thr.
__global__ void caps_tpass(const __bf16* __restrict__ WbB, const __bf16* __restrict__ XbB,
                           const float* __restrict__ vsum, float* __restrict__ tb) {
  const int j = blockIdx.x >> 4, chunk = blockIdx.x & 15;
  const int w = threadIdx.x >> 5, lane = threadIdx.x & 31;
  const int l = lane & 15, h = lane >> 4;
  float vr[2][2][8];
#pragma unroll
  for (int bt = 0; bt < 2; ++bt)
#pragma unroll
    for (int mt = 0; mt < 2; ++mt)
#pragma unroll
      for (int r = 0; r < 8; ++r)
        vr[bt][mt][r] = vsum[((size_t)(bt * 16 + l) * 64 + j) * 32 + mt * 16 + 8 * h + r];

  const v16bf zb = vzero16bf();
  const v8f   z8 = vzero8f();
  const int pbase = chunk * 64 + w * 16;
  for (int pp = 0; pp < 16; ++pp) {
    const int p = pbase + pp;
    const __bf16* wb = WbB + ((size_t)(j * 1024 + p) * 2) * 512 + (size_t)lane * 16;
    const v16bf A0 = *(const v16bf*)(wb);        // W^T as A operand
    const v16bf A1 = *(const v16bf*)(wb + 512);
#pragma unroll
    for (int bt = 0; bt < 2; ++bt) {
      const v16bf xf = *(const v16bf*)(XbB + ((size_t)(bt * 1024 + p)) * 512 + (size_t)lane * 16);
      const v16bf x0 = (h == 0) ? xf : zb;         // B-layout: lane half = i select
      const v16bf x1 = (h == 0) ? zb : xf;
      v8f D00 = WMMA_BF16(A0, x0, z8);
      v8f D10 = WMMA_BF16(A1, x0, z8);
      v8f D01 = WMMA_BF16(A0, x1, z8);
      v8f D11 = WMMA_BF16(A1, x1, z8);
      float t0 = 0.0f, t1 = 0.0f;
#pragma unroll
      for (int r = 0; r < 8; ++r) {
        t0 += D00[r] * vr[bt][0][r] + D10[r] * vr[bt][1][r];
        t1 += D01[r] * vr[bt][0][r] + D11[r] * vr[bt][1][r];
      }
      t0 += __shfl_xor(t0, 16, 32);                // combine m-halves across lane halves
      t1 += __shfl_xor(t1, 16, 32);
      if (lane < 16) {
        const int b = bt * 16 + lane;
        tb[((size_t)j * 2048 + 2 * p    ) * 32 + b] = t0;
        tb[((size_t)j * 2048 + 2 * p + 1) * 32 + b] = t1;
      }
    }
  }
}

// --- softmax over j for each (b,i); t,c layout [j][i][b] -> fully coalesced.
__global__ void caps_softmax(const float* __restrict__ tb, float* __restrict__ cb) {
  const int tid = blockIdx.x * blockDim.x + threadIdx.x;   // 65536 threads
  const int b = tid & 31, i = tid >> 5;
  float mx = -1e30f;
  for (int j = 0; j < 64; ++j) mx = fmaxf(mx, tb[((size_t)j * 2048 + i) * 32 + b]);
  float sum = 0.0f;
  for (int j = 0; j < 64; ++j) sum += __expf(tb[((size_t)j * 2048 + i) * 32 + b] - mx);
  const float inv = 1.0f / sum;
  for (int j = 0; j < 64; ++j)
    cb[((size_t)j * 2048 + i) * 32 + b] = __expf(tb[((size_t)j * 2048 + i) * 32 + b] - mx) * inv;
}

extern "C" void kernel_launch(void* const* d_in, const int* in_sizes, int n_in,
                              void* d_out, int out_size, void* d_ws, size_t ws_size,
                              hipStream_t stream) {
  (void)in_sizes; (void)n_in; (void)out_size;
  if (ws_size < WS_NEED) return;
  const float* x    = (const float*)d_in[0];
  const float* W    = (const float*)d_in[1];
  const float* bias = (const float*)d_in[2];
  float* out = (float*)d_out;
  char*  ws  = (char*)d_ws;
  __bf16* WbB = (__bf16*)(ws + OFF_WB);
  __bf16* XbA = (__bf16*)(ws + OFF_XA);
  __bf16* XbB = (__bf16*)(ws + OFF_XB);
  float*  tb  = (float*)(ws + OFF_T);
  float*  cb  = (float*)(ws + OFF_C);
  float*  sp  = (float*)(ws + OFF_SP);
  float*  vs  = (float*)(ws + OFF_VS);

  caps_convW<<<16384, 256, 0, stream>>>(W, WbB);   // 131072 fragment-waves
  caps_convX<<<8, 256, 0, stream>>>(x, XbA, XbB);  // 64 fragment-waves

  for (int it = 0; it < 3; ++it) {
    caps_spass<<<512, 256, 0, stream>>>(WbB, XbA, cb, it == 0 ? 1 : 0, sp);
    caps_squash<<<2048, 32, 0, stream>>>(sp, bias, out, vs, it == 0 ? 1 : 0);
    if (it < 2) {
      caps_tpass<<<1024, 128, 0, stream>>>(WbB, XbB, vs, tb);
      caps_softmax<<<256, 256, 0, stream>>>(tb, cb);
    }
  }
}
